// DecoderTransformerBlock_43130061586466
// MI455X (gfx1250) — matrix operand order repaired and checked by reading, not compiled, not run on hardware
//
#include <hip/hip_runtime.h>
#include <hip/hip_bf16.h>

typedef __bf16 bf16_t;
typedef __attribute__((ext_vector_type(16))) __bf16 v16bf;
typedef __attribute__((ext_vector_type(8)))  __bf16 v8bf;
typedef __attribute__((ext_vector_type(8)))  float  v8f;
typedef __attribute__((ext_vector_type(4))) unsigned int u32x4;
typedef __attribute__((ext_vector_type(8))) int i32x8;
typedef __attribute__((ext_vector_type(4))) int i32x4;

#define DEV __device__ __forceinline__

namespace {

constexpr int C        = 256;
constexpr int HEADS    = 8;
constexpr int HD       = 32;
constexpr int NWIN     = 2048;    // B * 32 * 32 windows
constexpr int NW_PER_B = 1024;
constexpr int NTOK     = 64;      // tokens per 8x8 window
constexpr int TOKENS   = 131072;  // 2 * 256 * 256
constexpr float SCALE  = 0.17677669529663689f;  // 32^-0.5

DEV v8f zero8() {
  v8f z;
#pragma unroll
  for (int i = 0; i < 8; ++i) z[i] = 0.0f;
  return z;
}

DEV v8f wmma_bf16(v16bf a, v16bf b, v8f c) {
  return __builtin_amdgcn_wmma_f32_16x16x32_bf16(false, a, false, b, (short)0, c,
                                                 false, false);
}

// A fragment 16x32 (MxK) from row-major source (leading dim ld, elements).
// Lane holds row=lane&15; lanes 0-15 elems 0..15 = K[k0..k0+15], lanes 16-31 K=+16.
DEV v16bf load_a(const bf16_t* base, int ld, int row0, int k0, int lane) {
  const int half = lane >> 4, m = lane & 15;
  const bf16_t* p = base + (size_t)(row0 + m) * ld + (k0 + half * 16);
  v8bf lo = *(const v8bf*)p;
  v8bf hi = *(const v8bf*)(p + 8);
  v16bf a;
#pragma unroll
  for (int i = 0; i < 8; ++i) { a[i] = lo[i]; a[i + 8] = hi[i]; }
  return a;
}

// B fragment 32x16 (KxN) where B[k][n] = W[n0+n][k0+k], W row-major (ld).
// Lane holds column n=lane&15; lanes 0-15 K=k0..k0+15, lanes 16-31 K=+16.
DEV v16bf load_b(const bf16_t* base, int ld, int n0, int k0, int lane) {
  const int half = lane >> 4, n = lane & 15;
  return *(const v16bf*)(base + (size_t)(n0 + n) * ld + (k0 + half * 16));
}

DEV float lane_reduce_add(float v) {
#pragma unroll
  for (int m = 1; m < 32; m <<= 1) v += __shfl_xor(v, m, 32);
  return v;
}

DEV float gelu_exact(float x) {
  return 0.5f * x * (1.0f + erff(x * 0.70710678118654752f));
}

// Tensor Data Mover: 1-D 32KB tile (64x256 bf16) global -> LDS.
// D# per CDNA5 ISA: data_size=8B, tensor_dim0=tile_dim0=4096,
// no multicast / pad / iterate / gather; groups 2-3 zero (tile_dim2/3 unused).
DEV void tdm_load_window(const bf16_t* gsrc, unsigned lds_byte_off) {
  const unsigned long long ga = (unsigned long long)(uintptr_t)gsrc;
  u32x4 g0;
  g0[0] = 1u;                                        // count=1 user descriptor
  g0[1] = lds_byte_off;                              // lds_addr
  g0[2] = (unsigned)(ga & 0xFFFFFFFFu);              // global_addr[31:0]
  g0[3] = (unsigned)((ga >> 32) & 0x1FFFFFFu) | (2u << 30);  // [56:32] | type=2
  i32x8 g1;
  g1[0] = (int)(3u << 16);        // workgroup_mask=0, data_size=3 (8B)
  g1[1] = (int)(4096u << 16);     // tensor_dim0[15:0] in bits [31:16]
  g1[2] = 0;                      // tensor_dim0[31:16]=0, tensor_dim1 lo=0
  g1[3] = (int)(4096u << 16);     // tile_dim0 = 4096
  g1[4] = 0;                      // tile_dim1 = 0 (unused), tile_dim2 = 0
  g1[5] = 4096;                   // tensor_dim0_stride lo
  g1[6] = 0;
  g1[7] = 0;
  i32x4 gz = {0, 0, 0, 0};
#if defined(__clang_major__) && (__clang_major__ >= 23)
  i32x8 gz8 = {0, 0, 0, 0, 0, 0, 0, 0};
  __builtin_amdgcn_tensor_load_to_lds(g0, g1, gz, gz, gz8, 0);
#else
  __builtin_amdgcn_tensor_load_to_lds(g0, g1, gz, gz, 0);
#endif
}

}  // namespace

__global__ __launch_bounds__(256) void k_cvt_bf16(const float* __restrict__ s,
                                                  bf16_t* __restrict__ d, int n) {
  int i = blockIdx.x * 256 + threadIdx.x;
  if (i < n) d[i] = (bf16_t)s[i];
}

// LayerNorm + shifted-window partition (roll by -4 folded into the gather).
// One window per block, wave-per-token (8 channels per lane).
__global__ __launch_bounds__(256) void k_ln_window(const float* __restrict__ src,
    const float* __restrict__ gamma, const float* __restrict__ beta,
    bf16_t* __restrict__ dst) {
  const int w = blockIdx.x;
  const int wave = threadIdx.x >> 5, lane = threadIdx.x & 31;
  const int b = w >> 10, wl = w & 1023, wi = wl >> 5, wj = wl & 31;
  float g[8], bt[8];
#pragma unroll
  for (int i = 0; i < 8; ++i) { g[i] = gamma[lane * 8 + i]; bt[i] = beta[lane * 8 + i]; }
  for (int it = 0; it < 8; ++it) {
    const int t = wave * 8 + it;
    const int row = (wi * 8 + (t >> 3) + 4) & 255;
    const int col = (wj * 8 + (t & 7) + 4) & 255;
    const float* p = src + (size_t)((b * 256 + row) * 256 + col) * C + lane * 8;
    float v[8];
    float s = 0.f, ss = 0.f;
#pragma unroll
    for (int i = 0; i < 8; ++i) { v[i] = p[i]; s += v[i]; ss += v[i] * v[i]; }
    s = lane_reduce_add(s);
    ss = lane_reduce_add(ss);
    const float mean = s * (1.0f / C);
    const float inv = rsqrtf(ss * (1.0f / C) - mean * mean + 1e-5f);
    bf16_t o[8];
#pragma unroll
    for (int i = 0; i < 8; ++i) o[i] = (bf16_t)((v[i] - mean) * inv * g[i] + bt[i]);
    *(v8bf*)(dst + ((size_t)w * NTOK + t) * C + lane * 8) = *(const v8bf*)o;
  }
}

// Fused windowed attention: TDM-stage activations -> QKV proj -> per-head
// softmax attn -> out proj -> reverse-roll residual scatter-add.
// One block (8 waves) per window. LDS 160KB -> 2 workgroups/WGP.
__global__ __launch_bounds__(256) void k_win_attn(
    const bf16_t* __restrict__ hq, const bf16_t* __restrict__ hkv,
    const bf16_t* __restrict__ qw, const float* __restrict__ qb,
    const bf16_t* __restrict__ kvw, const float* __restrict__ kvb,
    const bf16_t* __restrict__ pw, const float* __restrict__ pb,
    const float* __restrict__ tab, const float* __restrict__ maskp,
    const float* __restrict__ resid, float* __restrict__ dst) {
  __shared__ __align__(32) bf16_t sQ[NTOK * C];   // 32 KB, reused as O
  __shared__ __align__(32) bf16_t sK[NTOK * C];   // 32 KB
  __shared__ __align__(32) bf16_t sVt[C * NTOK];  // 32 KB, [channel][token]
  // 64 KB arena: phase 1 = TDM-staged hq|hkv windows; phase 2 = scores|probs.
  __shared__ __align__(32) unsigned char sArena[65536];

  const int w = blockIdx.x;
  const int wave = threadIdx.x >> 5, lane = threadIdx.x & 31;
  const int half = lane >> 4, lcol = lane & 15;
  const bf16_t* hq_w = hq + (size_t)w * NTOK * C;
  const bf16_t* hkv_w = hkv + (size_t)w * NTOK * C;
  const bool self_attn = (hq == hkv);

  bf16_t* sHq = (bf16_t*)sArena;
  bf16_t* sHk = self_attn ? sHq : (bf16_t*)(sArena + 32768);

  // ---- Phase 0: Tensor Data Mover stages window activations into LDS ----
  if (wave == 0) {
    tdm_load_window(hq_w, (unsigned)(uintptr_t)sHq);
    if (!self_attn) tdm_load_window(hkv_w, (unsigned)(uintptr_t)sHk);
    __builtin_amdgcn_s_wait_tensorcnt(0);
  }
  __syncthreads();

  // ---- Phase 1: Q / K / V projections (bf16 WMMA, fp32 accumulate) ----
  for (int tt = wave; tt < 192; tt += 8) {
    const int seg = tt >> 6;           // 0=Q 1=K 2=V
    const int loc = tt & 63;
    const int row0 = (loc >> 4) * 16;  // token tile
    const int n0 = (loc & 15) * 16;    // out-channel tile
    const bf16_t* src = (seg == 0) ? sHq : sHk;
    const bf16_t* wmat = (seg == 0) ? qw : (seg == 1 ? kvw : (kvw + C * C));
    v8f acc = zero8();
#pragma unroll
    for (int kk = 0; kk < 8; ++kk)
      acc = wmma_bf16(load_a(src, C, row0, kk * 32, lane),
                      load_b(wmat, C, n0, kk * 32, lane), acc);
    const int n = n0 + lcol;
    if (seg == 0) {
      const float bias = qb[n];
#pragma unroll
      for (int v = 0; v < 8; ++v) {
        const int tok = row0 + v + half * 8;
        sQ[tok * C + n] = (bf16_t)((acc[v] + bias) * SCALE);
      }
    } else if (seg == 1) {
      const float bias = kvb[n];
#pragma unroll
      for (int v = 0; v < 8; ++v) {
        const int tok = row0 + v + half * 8;
        sK[tok * C + n] = (bf16_t)(acc[v] + bias);
      }
    } else {
      const float bias = kvb[C + n];
#pragma unroll
      for (int v = 0; v < 8; ++v) {
        const int tok = row0 + v + half * 8;
        sVt[n * NTOK + tok] = (bf16_t)(acc[v] + bias);  // transposed for PV B-frags
      }
    }
  }
  __syncthreads();

  // Phase-2 view of the arena (phase-1 tiles are dead now).
  float* scBase = (float*)sArena;                 // 8 heads x 16x64 f32 scores
  bf16_t* pBase = (bf16_t*)(sArena + 32768);      // 8 heads x 16x64 bf16 probs

  // ---- Phase 2: one head per wave; 4 row-blocks of 16 queries ----
  const int h = wave;
  const int mw = w & (NW_PER_B - 1);
  for (int rb = 0; rb < 4; ++rb) {
    // scores S(16x64) = Q[rb](16x32) x K^T(32x64): hd=32 -> single K-step
#pragma unroll
    for (int ct = 0; ct < 4; ++ct) {
      v8f s = wmma_bf16(load_a(sQ, C, rb * 16, h * HD, lane),
                        load_b(sK, C, ct * 16, h * HD, lane), zero8());
#pragma unroll
      for (int v = 0; v < 8; ++v)
        scBase[h * 1024 + (v + half * 8) * 64 + ct * 16 + lcol] = s[v];
    }
    __syncthreads();
    // softmax with relative-position bias + shift mask (fp32)
    {
      const int lr = lane >> 1;
      const int coff = (lane & 1) * 32;
      const int qi = rb * 16 + lr;
      const int ri = qi >> 3, ci = qi & 7;
      float e[32];
      float mx = -3.0e38f;
#pragma unroll
      for (int j = 0; j < 32; ++j) {
        const int kj = coff + j;
        const int ridx = (ri - (kj >> 3) + 7) * 15 + (ci - (kj & 7) + 7);
        float sv = scBase[h * 1024 + lr * 64 + kj] + tab[ridx * HEADS + h] +
                   maskp[(size_t)mw * 4096 + qi * 64 + kj];
        e[j] = sv;
        mx = fmaxf(mx, sv);
      }
      mx = fmaxf(mx, __shfl_xor(mx, 1, 32));
      float sum = 0.f;
#pragma unroll
      for (int j = 0; j < 32; ++j) { e[j] = __expf(e[j] - mx); sum += e[j]; }
      sum += __shfl_xor(sum, 1, 32);
      const float inv = 1.0f / sum;
#pragma unroll
      for (int j = 0; j < 32; ++j)
        pBase[h * 1024 + lr * 64 + coff + j] = (bf16_t)(e[j] * inv);
    }
    __syncthreads();
    // O[rb](16x32) = P(16x64) x V_h(64x32); write over sQ (rows rb, cols h*32)
    v8f o0 = zero8(), o1 = zero8();
#pragma unroll
    for (int kk = 0; kk < 2; ++kk) {
      v16bf a = load_a(pBase + h * 1024, 64, 0, kk * 32, lane);
      o0 = wmma_bf16(a, load_b(sVt, NTOK, h * HD, kk * 32, lane), o0);
      o1 = wmma_bf16(a, load_b(sVt, NTOK, h * HD + 16, kk * 32, lane), o1);
    }
#pragma unroll
    for (int v = 0; v < 8; ++v) {
      const int tok = rb * 16 + v + half * 8;
      sQ[tok * C + h * HD + lcol] = (bf16_t)o0[v];
      sQ[tok * C + h * HD + 16 + lcol] = (bf16_t)o1[v];
    }
  }
  __syncthreads();

  // ---- Phase 3: output projection + reverse-roll residual scatter-add ----
  const int b = w >> 10, wl = w & 1023, wi = wl >> 5, wj = wl & 31;
  for (int t = wave; t < 64; t += 8) {
    const int row0 = (t >> 4) * 16;
    const int n0 = (t & 15) * 16;
    v8f acc = zero8();
#pragma unroll
    for (int kk = 0; kk < 8; ++kk)
      acc = wmma_bf16(load_a(sQ, C, row0, kk * 32, lane),
                      load_b(pw, C, n0, kk * 32, lane), acc);
    const int n = n0 + lcol;
    const float bias = pb[n];
#pragma unroll
    for (int v = 0; v < 8; ++v) {
      const int tok = row0 + v + half * 8;
      const int row = (wi * 8 + (tok >> 3) + 4) & 255;
      const int col = (wj * 8 + (tok & 7) + 4) & 255;
      const size_t idx = (size_t)((b * 256 + row) * 256 + col) * C + n;
      dst[idx] = resid[idx] + acc[v] + bias;
    }
  }
}

// Fused LN3 -> fc1 -> GELU -> fc2 -> residual. 64 tokens per block; hidden
// processed in two 512-wide halves staged in LDS; fc2 accumulated in VGPRs.
__global__ __launch_bounds__(256) void k_mlp(
    const float* __restrict__ src,
    const float* __restrict__ n3w, const float* __restrict__ n3b,
    const bf16_t* __restrict__ fc1w, const float* __restrict__ fc1b,
    const bf16_t* __restrict__ fc2w, const float* __restrict__ fc2b,
    const float* __restrict__ resid, float* __restrict__ out) {
  __shared__ __align__(32) bf16_t sX[NTOK * C];     // 32 KB
  __shared__ __align__(32) bf16_t sHd[NTOK * 512];  // 64 KB (half of hidden)
  const int wave = threadIdx.x >> 5, lane = threadIdx.x & 31;
  const int half = lane >> 4, lcol = lane & 15;
  const size_t tok0 = (size_t)blockIdx.x * NTOK;

  // LayerNorm into sX
  for (int it = 0; it < 8; ++it) {
    const int t = wave * 8 + it;
    const float* p = src + (tok0 + t) * C + lane * 8;
    float v[8];
    float s = 0.f, ss = 0.f;
#pragma unroll
    for (int i = 0; i < 8; ++i) { v[i] = p[i]; s += v[i]; ss += v[i] * v[i]; }
    s = lane_reduce_add(s);
    ss = lane_reduce_add(ss);
    const float mean = s * (1.0f / C);
    const float inv = rsqrtf(ss * (1.0f / C) - mean * mean + 1e-5f);
    bf16_t o[8];
#pragma unroll
    for (int i = 0; i < 8; ++i) {
      const int ch = lane * 8 + i;
      o[i] = (bf16_t)((v[i] - mean) * inv * n3w[ch] + n3b[ch]);
    }
    *(v8bf*)(&sX[t * C + lane * 8]) = *(const v8bf*)o;
  }
  __syncthreads();

  v8f acc[8];
#pragma unroll
  for (int i = 0; i < 8; ++i) acc[i] = zero8();

  for (int hh = 0; hh < 2; ++hh) {
    // fc1 + GELU for 512 hidden channels into sHd
    for (int t = wave; t < 128; t += 8) {
      const int row0 = (t >> 5) * 16;
      const int n0loc = (t & 31) * 16;
      const int n0 = hh * 512 + n0loc;
      v8f a1 = zero8();
#pragma unroll
      for (int kk = 0; kk < 8; ++kk)
        a1 = wmma_bf16(load_a(sX, C, row0, kk * 32, lane),
                       load_b(fc1w, C, n0, kk * 32, lane), a1);
      const float bias = fc1b[n0 + lcol];
#pragma unroll
      for (int v = 0; v < 8; ++v) {
        const int tok = row0 + v + half * 8;
        sHd[tok * 512 + n0loc + lcol] = (bf16_t)gelu_exact(a1[v] + bias);
      }
    }
    __syncthreads();
    // fc2 partial accumulation over this hidden half
#pragma unroll
    for (int i = 0; i < 8; ++i) {
      const int ot = wave + i * 8;
      const int row0 = (ot >> 4) * 16;
      const int n0 = (ot & 15) * 16;
#pragma unroll
      for (int kk = 0; kk < 16; ++kk)
        acc[i] = wmma_bf16(load_a(sHd, 512, row0, kk * 32, lane),
                           load_b(fc2w, 1024, n0, hh * 512 + kk * 32, lane), acc[i]);
    }
    __syncthreads();
  }

  // out = resid + fc2 + bias
#pragma unroll
  for (int i = 0; i < 8; ++i) {
    const int ot = wave + i * 8;
    const int row0 = (ot >> 4) * 16;
    const int n0 = (ot & 15) * 16;
    const float bias = fc2b[n0 + lcol];
#pragma unroll
    for (int v = 0; v < 8; ++v) {
      const size_t idx = (tok0 + row0 + v + half * 8) * C + n0 + lcol;
      out[idx] = resid[idx] + acc[i][v] + bias;
    }
  }
}

extern "C" void kernel_launch(void* const* d_in, const int* in_sizes, int n_in,
                              void* d_out, int out_size, void* d_ws, size_t ws_size,
                              hipStream_t stream) {
  (void)in_sizes; (void)n_in; (void)out_size; (void)ws_size;
  const float* x       = (const float*)d_in[0];
  const float* attn_kv = (const float*)d_in[1];
  const float* mask_q  = (const float*)d_in[2];
  const float* mask_kv = (const float*)d_in[3];
  const float* n1w = (const float*)d_in[4],  *n1b = (const float*)d_in[5];
  const float* n2w = (const float*)d_in[6],  *n2b = (const float*)d_in[7];
  const float* n3w = (const float*)d_in[8],  *n3b = (const float*)d_in[9];
  const float* nkvw = (const float*)d_in[10], *nkvb = (const float*)d_in[11];
  const float* a1_qw = (const float*)d_in[12], *a1_qb = (const float*)d_in[13];
  const float* a1_kvw = (const float*)d_in[14], *a1_kvb = (const float*)d_in[15];
  const float* a1_pw = (const float*)d_in[16], *a1_pb = (const float*)d_in[17];
  const float* a1_tab = (const float*)d_in[18];
  const float* a2_qw = (const float*)d_in[19], *a2_qb = (const float*)d_in[20];
  const float* a2_kvw = (const float*)d_in[21], *a2_kvb = (const float*)d_in[22];
  const float* a2_pw = (const float*)d_in[23], *a2_pb = (const float*)d_in[24];
  const float* a2_tab = (const float*)d_in[25];
  const float* fc1w = (const float*)d_in[26], *fc1b = (const float*)d_in[27];
  const float* fc2w = (const float*)d_in[28], *fc2b = (const float*)d_in[29];

  char* base = (char*)d_ws;
  size_t off = 0;
  auto take = [&](size_t bytes) -> void* {
    void* p = base + off;
    off += (bytes + 255) & ~(size_t)255;
    return p;
  };
  bf16_t* wq1 = (bf16_t*)take(65536 * sizeof(bf16_t));
  bf16_t* wkv1 = (bf16_t*)take(131072 * sizeof(bf16_t));
  bf16_t* wp1 = (bf16_t*)take(65536 * sizeof(bf16_t));
  bf16_t* wq2 = (bf16_t*)take(65536 * sizeof(bf16_t));
  bf16_t* wkv2 = (bf16_t*)take(131072 * sizeof(bf16_t));
  bf16_t* wp2 = (bf16_t*)take(65536 * sizeof(bf16_t));
  bf16_t* wf1 = (bf16_t*)take(262144 * sizeof(bf16_t));
  bf16_t* wf2 = (bf16_t*)take(262144 * sizeof(bf16_t));
  bf16_t* hWin1 = (bf16_t*)take((size_t)NWIN * NTOK * C * sizeof(bf16_t));
  bf16_t* hWin2 = (bf16_t*)take((size_t)NWIN * NTOK * C * sizeof(bf16_t));
  float* bufX = (float*)take((size_t)TOKENS * C * sizeof(float));

  auto cvt = [&](const float* s, bf16_t* d, int n) {
    k_cvt_bf16<<<dim3((n + 255) / 256), dim3(256), 0, stream>>>(s, d, n);
  };
  cvt(a1_qw, wq1, 65536);
  cvt(a1_kvw, wkv1, 131072);
  cvt(a1_pw, wp1, 65536);
  cvt(a2_qw, wq2, 65536);
  cvt(a2_kvw, wkv2, 131072);
  cvt(a2_pw, wp2, 65536);
  cvt(fc1w, wf1, 262144);
  cvt(fc2w, wf2, 262144);

  // shifted-window self-attention
  k_ln_window<<<dim3(NWIN), dim3(256), 0, stream>>>(x, n1w, n1b, hWin1);
  k_win_attn<<<dim3(NWIN), dim3(256), 0, stream>>>(hWin1, hWin1, wq1, a1_qb,
      wkv1, a1_kvb, wp1, a1_pb, a1_tab, mask_q, x, bufX);
  // shifted-window cross-attention
  k_ln_window<<<dim3(NWIN), dim3(256), 0, stream>>>(bufX, n2w, n2b, hWin1);
  k_ln_window<<<dim3(NWIN), dim3(256), 0, stream>>>(attn_kv, nkvw, nkvb, hWin2);
  k_win_attn<<<dim3(NWIN), dim3(256), 0, stream>>>(hWin1, hWin2, wq2, a2_qb,
      wkv2, a2_kvb, wp2, a2_pb, a2_tab, mask_kv, bufX, bufX);
  // MLP
  k_mlp<<<dim3(TOKENS / NTOK), dim3(256), 0, stream>>>(bufX, n3w, n3b, wf1, fc1b,
      wf2, fc2b, bufX, (float*)d_out);
}
